// MMFed_51445118271743
// MI455X (gfx1250) — compile-verified, matrix operand order checked by reading, not run on hardware
//
#include <hip/hip_runtime.h>

// ---------------------------------------------------------------------------
// MI455X (gfx1250): dense GEMMs via v_wmma_f32_16x16x32_bf16, wave32.
// Each wave computes a 32x32 output region (4 WMMAs / K-step, 2x fragment
// reuse); 8 waves/block -> 64x128 block tile.
// ---------------------------------------------------------------------------

typedef __attribute__((ext_vector_type(16))) __bf16        v16bf;
typedef __attribute__((ext_vector_type(8)))  float         v8f;
typedef __attribute__((ext_vector_type(4)))  unsigned int  u32x4;

#define B_    32
#define E_    512
#define H_    8
#define L_    4
#define T_    196
#define ROWS  6272          // B_*T_
#define PD_   3840
#define NCLS_ 101

__device__ __forceinline__ unsigned short f2bf(float f) {
  unsigned u = __float_as_uint(f);
  u += 0x7FFFu + ((u >> 16) & 1u);   // round-to-nearest-even
  return (unsigned short)(u >> 16);
}

// ---- f32 -> bf16 convert (grid-stride) ----
__global__ void k_f2bf(const float* __restrict__ in, unsigned short* __restrict__ out, long n) {
  long i = (long)blockIdx.x * blockDim.x + threadIdx.x;
  long stride = (long)gridDim.x * blockDim.x;
  for (; i < n; i += stride) out[i] = f2bf(in[i]);
}

__global__ void k_zero_ush(unsigned short* __restrict__ p, long n) {
  long i = (long)blockIdx.x * blockDim.x + threadIdx.x;
  long stride = (long)gridDim.x * blockDim.x;
  for (; i < n; i += stride) p[i] = 0;
}

// ---- Shifted Patch Tokenization value (recomputed, avoids 96MB f32 buffer) ----
__device__ __forceinline__ float patch_val(const float* __restrict__ img, int b, int t, int d) {
  int c  = d % 15;          // channel in concat([img,s1,s2,s3,s4])
  int pp = d / 15;
  int p2 = pp & 15;
  int p1 = pp >> 4;
  int hh = t / 14, ww = t % 14;
  int Hp = hh * 16 + p1, Wp = ww * 16 + p2;
  int ch = c % 3, sh = c / 3;
  if      (sh == 1) Wp -= 1;
  else if (sh == 2) Wp += 1;
  else if (sh == 3) Hp -= 1;
  else if (sh == 4) Hp += 1;
  if ((unsigned)Hp >= 224u || (unsigned)Wp >= 224u) return 0.f;
  return img[(((long)b * 3 + ch) * 224 + Hp) * 224 + Wp];
}

// one block per token row: LN over 3840 dims, emit bf16 patches
__global__ void k_spt_ln(const float* __restrict__ img, const float* __restrict__ g,
                         const float* __restrict__ bb, unsigned short* __restrict__ out) {
  int row = blockIdx.x;                 // b*196 + t
  int b = row / T_, t = row % T_;
  int tid = threadIdx.x;
  __shared__ float s0[256], s1[256];
  float sum = 0.f, ss = 0.f;
  for (int d = tid; d < PD_; d += 256) {
    float v = patch_val(img, b, t, d);
    sum += v; ss += v * v;
  }
  s0[tid] = sum; s1[tid] = ss; __syncthreads();
  for (int st = 128; st > 0; st >>= 1) {
    if (tid < st) { s0[tid] += s0[tid + st]; s1[tid] += s1[tid + st]; }
    __syncthreads();
  }
  float mean = s0[0] * (1.f / PD_);
  float var  = s1[0] * (1.f / PD_) - mean * mean;
  float inv  = rsqrtf(var + 1e-5f);
  unsigned short* orow = out + (long)row * PD_;
  for (int d = tid; d < PD_; d += 256) {
    float v = patch_val(img, b, t, d);
    orow[d] = f2bf((v - mean) * inv * g[d] + bb[d]);
  }
}

// LayerNorm over E=512, bf16 output. One block (256 thr) per row.
__global__ void k_ln512(const float* __restrict__ X, const float* __restrict__ g,
                        const float* __restrict__ bb, unsigned short* __restrict__ Y) {
  int row = blockIdx.x;
  int tid = threadIdx.x;
  const float* x = X + (long)row * E_;
  float v0 = x[tid], v1 = x[tid + 256];
  __shared__ float s0[256], s1[256];
  s0[tid] = v0 + v1; s1[tid] = v0 * v0 + v1 * v1;
  __syncthreads();
  for (int st = 128; st > 0; st >>= 1) {
    if (tid < st) { s0[tid] += s0[tid + st]; s1[tid] += s1[tid + st]; }
    __syncthreads();
  }
  float mean = s0[0] * (1.f / E_);
  float var  = s1[0] * (1.f / E_) - mean * mean;
  float inv  = rsqrtf(var + 1e-5f);
  unsigned short* y = Y + (long)row * E_;
  y[tid]       = f2bf((v0 - mean) * inv * g[tid]       + bb[tid]);
  y[tid + 256] = f2bf((v1 - mean) * inv * g[tid + 256] + bb[tid + 256]);
}

// Final LN on last tokens -> feat bf16 (32 x 1024). 64 blocks: [stream, batch].
__global__ void k_final_feat(const float* __restrict__ x1, const float* __restrict__ x2,
                             const float* __restrict__ g, const float* __restrict__ bb,
                             unsigned short* __restrict__ feat) {
  int blk = blockIdx.x;               // 0..63
  int s = blk >> 5, b = blk & 31;
  int tid = threadIdx.x;
  const float* X = (s ? x2 : x1) + ((long)b * T_ + (T_ - 1)) * E_;
  float v0 = X[tid], v1 = X[tid + 256];
  __shared__ float s0[256], s1[256];
  s0[tid] = v0 + v1; s1[tid] = v0 * v0 + v1 * v1;
  __syncthreads();
  for (int st = 128; st > 0; st >>= 1) {
    if (tid < st) { s0[tid] += s0[tid + st]; s1[tid] += s1[tid + st]; }
    __syncthreads();
  }
  float mean = s0[0] * (1.f / E_);
  float var  = s1[0] * (1.f / E_) - mean * mean;
  float inv  = rsqrtf(var + 1e-5f);
  unsigned short* y = feat + (long)b * 1024 + s * 512;
  y[tid]       = f2bf((v0 - mean) * inv * g[tid]       + bb[tid]);
  y[tid + 256] = f2bf((v1 - mean) * inv * g[tid + 256] + bb[tid + 256]);
}

// sinusoidal positional add (faithful to reference bug: indexed by batch)
__global__ void k_add_pos(float* __restrict__ x1, float* __restrict__ x2) {
  long n = (long)ROWS * E_;
  long i = (long)blockIdx.x * blockDim.x + threadIdx.x;
  long stride = (long)gridDim.x * blockDim.x;
  for (; i < n; i += stride) {
    int e = (int)(i % E_);
    int b = (int)(i / ((long)T_ * E_));
    int idx = (e < 256) ? e : e - 256;
    float freq = expf((float)idx * -0.03611898185f);     // -ln(10000)/255
    float ang  = (float)(b + 1) * freq;
    float p    = (e < 256) ? sinf(ang) : cosf(ang);
    x1[i] += p; x2[i] += p;
  }
}

// Fused attention: one block per (b, h, q-row). q/k/v live in qkv[b,n,3E].
__global__ void k_attention(const float* __restrict__ qkv, float* __restrict__ O) {
  int blk  = blockIdx.x;              // B_*H_*T_ blocks
  int qrow = blk % T_;
  int bh   = blk / T_;
  int h    = bh % H_;
  int b    = bh / H_;
  const float* base = qkv + (long)b * T_ * 1536;
  __shared__ float qv[64], sc[T_], red[256];
  int tid = threadIdx.x;
  if (tid < 64) qv[tid] = base[(long)qrow * 1536 + h * 64 + tid];
  __syncthreads();
  for (int k = tid; k < T_; k += 256) {
    const float* kp = base + (long)k * 1536 + 512 + h * 64;
    float s = 0.f;
    #pragma unroll 8
    for (int d = 0; d < 64; ++d) s += qv[d] * kp[d];
    sc[k] = s * 0.125f;               // 1/sqrt(64)
  }
  __syncthreads();
  float m = -1e30f;
  for (int k = tid; k < T_; k += 256) m = fmaxf(m, sc[k]);
  red[tid] = m; __syncthreads();
  for (int st = 128; st > 0; st >>= 1) { if (tid < st) red[tid] = fmaxf(red[tid], red[tid + st]); __syncthreads(); }
  float mx = red[0];
  __syncthreads();
  float sum = 0.f;
  for (int k = tid; k < T_; k += 256) { float e = expf(sc[k] - mx); sc[k] = e; sum += e; }
  __syncthreads();
  red[tid] = sum; __syncthreads();
  for (int st = 128; st > 0; st >>= 1) { if (tid < st) red[tid] += red[tid + st]; __syncthreads(); }
  float inv = 1.f / red[0];
  __syncthreads();
  if (tid < 64) {
    float o = 0.f;
    const float* vp = base + 1024 + h * 64 + tid;
    for (int k = 0; k < T_; ++k) o += sc[k] * vp[(long)k * 1536];
    O[((long)b * T_ + qrow) * E_ + h * 64 + tid] = o * inv;
  }
}

// ---------------------------------------------------------------------------
// WMMA GEMM:  C[M,N] (+)= A[M,K](bf16) @ W[N,K]^T(bf16) + bias
// 256 threads = 8 waves (2M x 4N); each wave owns a 32x32 region -> 4 WMMAs
// per 32-wide K step with 2x reuse of each fragment. Block tile: 64 x 128.
// K multiple of 32. flags: bit0 = relu, bit1 = accumulate into C.
// Fragment layout per ISA 7.12.2 (16-bit A/B): lane = r + 16*half,
// VGPR0..3 hold K = half*8+0..7, VGPR4..7 hold K = 16+half*8+0..7.
// ---------------------------------------------------------------------------
union FragBF { v16bf v; u32x4 q[2]; };

__device__ __forceinline__ void store_tile(float* __restrict__ C, int ldc,
                                           const v8f& acc, int row, int col,
                                           int M, int N, const float* __restrict__ bias,
                                           int flags) {
  if (col >= N) return;
  float bv = bias ? bias[col] : 0.f;
  #pragma unroll
  for (int i = 0; i < 8; ++i) {
    int rr = row + i;
    if (rr < M) {
      float v = acc[i] + bv;
      if (flags & 1) v = fmaxf(v, 0.f);
      float* p = C + (long)rr * ldc + col;
      if (flags & 2) *p += v; else *p = v;
    }
  }
}

__global__ void k_gemm_bf16(const unsigned short* __restrict__ A, int lda,
                            const unsigned short* __restrict__ W, int ldw,
                            const float* __restrict__ bias,
                            float* __restrict__ C, int ldc,
                            int M, int N, int K, int flags) {
  int tid  = threadIdx.x;
  int wave = tid >> 5;            // wave32
  int lane = tid & 31;
  int wm = wave >> 2;             // 0..1
  int wn = wave & 3;              // 0..3
  int row0 = blockIdx.y * 64  + wm * 32;   // wave rows  [row0, row0+32)
  int col0 = blockIdx.x * 128 + wn * 32;   // wave cols  [col0, col0+32)
  int r    = lane & 15;
  int half = lane >> 4;

  int ar0 = row0 + r;       if (ar0 >= M) ar0 = M - 1;   // clamp (stores guarded)
  int ar1 = row0 + 16 + r;  if (ar1 >= M) ar1 = M - 1;
  const unsigned short* pA0 = A + (long)ar0 * lda + half * 8;
  const unsigned short* pA1 = A + (long)ar1 * lda + half * 8;
  const unsigned short* pW0 = W + (long)(col0 + r)      * ldw + half * 8;
  const unsigned short* pW1 = W + (long)(col0 + 16 + r) * ldw + half * 8;

  v8f acc00 = {}, acc01 = {}, acc10 = {}, acc11 = {};
  for (int kb = 0; kb < K; kb += 32) {
    FragBF a0, a1, b0, b1;
    a0.q[0] = *(const u32x4*)(pA0 + kb);  a0.q[1] = *(const u32x4*)(pA0 + kb + 16);
    a1.q[0] = *(const u32x4*)(pA1 + kb);  a1.q[1] = *(const u32x4*)(pA1 + kb + 16);
    b0.q[0] = *(const u32x4*)(pW0 + kb);  b0.q[1] = *(const u32x4*)(pW0 + kb + 16);
    b1.q[0] = *(const u32x4*)(pW1 + kb);  b1.q[1] = *(const u32x4*)(pW1 + kb + 16);
    if (kb + 64 < K) {                    // emits global_prefetch_b8
      __builtin_prefetch(pA0 + kb + 64, 0, 1);
      __builtin_prefetch(pA1 + kb + 64, 0, 1);
      __builtin_prefetch(pW0 + kb + 64, 0, 1);
      __builtin_prefetch(pW1 + kb + 64, 0, 1);
    }
    acc00 = __builtin_amdgcn_wmma_f32_16x16x32_bf16(false, a0.v, false, b0.v, (short)0, acc00, false, false);
    acc01 = __builtin_amdgcn_wmma_f32_16x16x32_bf16(false, a0.v, false, b1.v, (short)0, acc01, false, false);
    acc10 = __builtin_amdgcn_wmma_f32_16x16x32_bf16(false, a1.v, false, b0.v, (short)0, acc10, false, false);
    acc11 = __builtin_amdgcn_wmma_f32_16x16x32_bf16(false, a1.v, false, b1.v, (short)0, acc11, false, false);
  }

  int colA = col0 + (lane & 15);          // C layout: VGPR i -> row (half*8 + i)
  int rb   = half * 8;
  store_tile(C, ldc, acc00, row0 + rb,      colA,      M, N, bias, flags);
  store_tile(C, ldc, acc01, row0 + rb,      colA + 16, M, N, bias, flags);
  store_tile(C, ldc, acc10, row0 + 16 + rb, colA,      M, N, bias, flags);
  store_tile(C, ldc, acc11, row0 + 16 + rb, colA + 16, M, N, bias, flags);
}

// ---------------------------------------------------------------------------
extern "C" void kernel_launch(void* const* d_in, const int* in_sizes, int n_in,
                              void* d_out, int out_size, void* d_ws, size_t ws_size,
                              hipStream_t stream) {
  (void)in_sizes; (void)n_in; (void)out_size; (void)ws_size;

  const float* image       = (const float*)d_in[0];
  const float* text_hidden = (const float*)d_in[1];
  const float* patch_ln_g  = (const float*)d_in[2];
  const float* patch_ln_b  = (const float*)d_in[3];
  const float* patch_w     = (const float*)d_in[4];
  const float* patch_b     = (const float*)d_in[5];
  const float* text_w      = (const float*)d_in[6];
  const float* text_b      = (const float*)d_in[7];
  const float* attn_in_w   = (const float*)d_in[8];
  const float* attn_in_b   = (const float*)d_in[9];
  const float* attn_out_w  = (const float*)d_in[10];
  const float* attn_out_b  = (const float*)d_in[11];
  const float* mlp_w1      = (const float*)d_in[12];
  const float* mlp_b1      = (const float*)d_in[13];
  const float* mlp_w2      = (const float*)d_in[14];
  const float* mlp_b2      = (const float*)d_in[15];
  const float* ln_g        = (const float*)d_in[16];
  const float* ln_b        = (const float*)d_in[17];
  const float* final_ln_g  = (const float*)d_in[18];
  const float* final_ln_b  = (const float*)d_in[19];
  const float* cls_w1      = (const float*)d_in[20];
  const float* cls_b1      = (const float*)d_in[21];
  const float* cls_w2      = (const float*)d_in[22];
  const float* cls_b2      = (const float*)d_in[23];
  float* out = (float*)d_out;

  // ---- workspace layout (256B aligned slabs) ----
  char* ws = (char*)d_ws;
  size_t off = 0;
  auto ALLOC = [&](size_t bytes) -> void* {
    void* p = ws + off;
    off = (off + bytes + 255) & ~(size_t)255;
    return p;
  };
  unsigned short* w_patch = (unsigned short*)ALLOC((size_t)512 * 3840 * 2);
  unsigned short* w_text  = (unsigned short*)ALLOC((size_t)512 * 768 * 2);
  unsigned short* w_ain   = (unsigned short*)ALLOC((size_t)L_ * 3 * 1536 * 512 * 2);
  unsigned short* w_aout  = (unsigned short*)ALLOC((size_t)L_ * 3 * 512 * 512 * 2);
  unsigned short* w_m1    = (unsigned short*)ALLOC((size_t)L_ * 2048 * 512 * 2);
  unsigned short* w_m2    = (unsigned short*)ALLOC((size_t)L_ * 512 * 2048 * 2);
  unsigned short* w_c1    = (unsigned short*)ALLOC((size_t)2048 * 1024 * 2);
  unsigned short* w_c2    = (unsigned short*)ALLOC((size_t)128 * 2048 * 2);   // padded 101->128
  unsigned short* patches = (unsigned short*)ALLOC((size_t)ROWS * PD_ * 2);
  float*          x1      = (float*)ALLOC((size_t)ROWS * E_ * 4);
  float*          x2      = (float*)ALLOC((size_t)ROWS * E_ * 4);
  unsigned short* ln_bf   = (unsigned short*)ALLOC((size_t)ROWS * E_ * 2);
  unsigned short* act_bf  = (unsigned short*)ALLOC((size_t)ROWS * 2048 * 2);
  float*          qkv     = (float*)ALLOC((size_t)ROWS * 1536 * 4);
  float*          obuf    = (float*)ALLOC((size_t)ROWS * E_ * 4);
  float*          mlph    = (float*)ALLOC((size_t)ROWS * 2048 * 4);
  unsigned short* feat_bf = (unsigned short*)ALLOC((size_t)32 * 1024 * 2);
  float*          clsh    = (float*)ALLOC((size_t)32 * 2048 * 4);
  unsigned short* clsh_bf = (unsigned short*)ALLOC((size_t)32 * 2048 * 2);

  auto CONV = [&](const float* src, unsigned short* dst, long n) {
    int blocks = (int)((n + 255) / 256);
    if (blocks > 4096) blocks = 4096;
    k_f2bf<<<blocks, 256, 0, stream>>>(src, dst, n);
  };
  auto GEMM = [&](const unsigned short* A, int lda, const unsigned short* Wp, int ldw,
                  const float* bias, float* C, int ldc, int M, int N, int K, int flags) {
    dim3 grid((unsigned)((N + 127) / 128), (unsigned)((M + 63) / 64));
    k_gemm_bf16<<<grid, 256, 0, stream>>>(A, lda, Wp, ldw, bias, C, ldc, M, N, K, flags);
  };

  // ---- weights -> bf16 ----
  CONV(patch_w,    w_patch, (long)512 * 3840);
  CONV(text_w,     w_text,  (long)512 * 768);
  CONV(attn_in_w,  w_ain,   (long)L_ * 3 * 1536 * 512);
  CONV(attn_out_w, w_aout,  (long)L_ * 3 * 512 * 512);
  CONV(mlp_w1,     w_m1,    (long)L_ * 2048 * 512);
  CONV(mlp_w2,     w_m2,    (long)L_ * 512 * 2048);
  CONV(cls_w1,     w_c1,    (long)2048 * 1024);
  k_zero_ush<<<256, 256, 0, stream>>>(w_c2, (long)128 * 2048);
  CONV(cls_w2,     w_c2,    (long)NCLS_ * 2048);

  // ---- embeddings ----
  k_spt_ln<<<ROWS, 256, 0, stream>>>(image, patch_ln_g, patch_ln_b, patches);
  GEMM(patches, PD_, w_patch, PD_, patch_b, x1, E_, ROWS, E_, PD_, 0);
  CONV(text_hidden, act_bf, (long)ROWS * 768);
  GEMM(act_bf, 768, w_text, 768, text_b, x2, E_, ROWS, E_, 768, 0);
  k_add_pos<<<2048, 256, 0, stream>>>(x1, x2);

  // ---- transformer layers ----
  for (int l = 0; l < L_; ++l) {
    const float* g0 = ln_g + (long)(l * 2 + 0) * E_;
    const float* b0 = ln_b + (long)(l * 2 + 0) * E_;
    const float* g1 = ln_g + (long)(l * 2 + 1) * E_;
    const float* b1 = ln_b + (long)(l * 2 + 1) * E_;

    // self-attention, both streams (residual add fused via accumulate flag)
    for (int s = 0; s < 2; ++s) {
      float* xs = s ? x2 : x1;
      const unsigned short* Wqkv = w_ain  + (long)(l * 3 + s) * 1536 * 512;
      const float*          Bqkv = attn_in_b  + (long)(l * 3 + s) * 1536;
      const unsigned short* Wout = w_aout + (long)(l * 3 + s) * 512 * 512;
      const float*          Bout = attn_out_b + (long)(l * 3 + s) * 512;
      k_ln512<<<ROWS, 256, 0, stream>>>(xs, g0, b0, ln_bf);
      GEMM(ln_bf, E_, Wqkv, E_, Bqkv, qkv, 1536, ROWS, 1536, E_, 0);
      k_attention<<<B_ * H_ * T_, 256, 0, stream>>>(qkv, obuf);
      CONV(obuf, act_bf, (long)ROWS * E_);
      GEMM(act_bf, E_, Wout, E_, Bout, xs, E_, ROWS, E_, E_, 2 /*acc: residual*/);
    }

    // guide attention: q from x2, k/v from x1; output REPLACES x1 (no residual)
    {
      const unsigned short* Wg = w_ain + (long)(l * 3 + 2) * 1536 * 512;
      const float*          Bg = attn_in_b + (long)(l * 3 + 2) * 1536;
      CONV(x2, ln_bf,  (long)ROWS * E_);       // query source (raw)
      CONV(x1, act_bf, (long)ROWS * E_);       // key/value source (raw)
      GEMM(ln_bf,  E_, Wg,                   E_, Bg,       qkv,       1536, ROWS, 512,  E_, 0);
      GEMM(act_bf, E_, Wg + (long)512 * 512, E_, Bg + 512, qkv + 512, 1536, ROWS, 1024, E_, 0);
      k_attention<<<B_ * H_ * T_, 256, 0, stream>>>(qkv, obuf);
      CONV(obuf, act_bf, (long)ROWS * E_);
      const unsigned short* Wout = w_aout + (long)(l * 3 + 2) * 512 * 512;
      const float*          Bout = attn_out_b + (long)(l * 3 + 2) * 512;
      GEMM(act_bf, E_, Wout, E_, Bout, x1, E_, ROWS, E_, E_, 0 /*overwrite*/);
    }

    // MLP (shared weights across streams), residual fused
    for (int s = 0; s < 2; ++s) {
      float* xs = s ? x2 : x1;
      const unsigned short* W1  = w_m1 + (long)l * 2048 * 512;
      const float*          Bb1 = mlp_b1 + (long)l * 2048;
      const unsigned short* W2  = w_m2 + (long)l * 512 * 2048;
      const float*          Bb2 = mlp_b2 + (long)l * 512;
      k_ln512<<<ROWS, 256, 0, stream>>>(xs, g1, b1, ln_bf);
      GEMM(ln_bf, E_, W1, E_, Bb1, mlph, 2048, ROWS, 2048, E_, 1 /*relu*/);
      CONV(mlph, act_bf, (long)ROWS * 2048);
      GEMM(act_bf, 2048, W2, 2048, Bb2, xs, E_, ROWS, E_, 2048, 2 /*acc*/);
    }
  }

  // ---- head ----
  k_final_feat<<<64, 256, 0, stream>>>(x1, x2, final_ln_g, final_ln_b, feat_bf);
  GEMM(feat_bf, 1024, w_c1, 1024, cls_b1, clsh, 2048, 32, 2048, 1024, 1 /*relu*/);
  CONV(clsh, clsh_bf, (long)32 * 2048);
  GEMM(clsh_bf, 2048, w_c2, 2048, cls_b2, out, NCLS_, 32, NCLS_, 2048, 0);
}
